// GraphEncoder_63264868270681
// MI455X (gfx1250) — matrix operand order repaired and checked by reading, not compiled, not run on hardware
//
#include <hip/hip_runtime.h>
#include <hip/hip_bf16.h>

typedef __attribute__((ext_vector_type(16))) __bf16 v16bf;
typedef __attribute__((ext_vector_type(8)))  float  v8f;

#define GCOUNT 3200   // B*L graphs
#define E_N    50     // nodes per ego-graph
#define D_DIM  100    // embedding dim (K), padded to 128 (4 x K=32)
#define H_DIM  100    // head dim (N), padded to 112 (7 x N=16)
#define NT     7
#define KT     4
#define XS     128    // bf16 x row stride (16B-aligned fragment chunks)
#define KVS    113    // f32 k/v row stride (odd -> no LDS bank conflicts)

// ---- one-time weight re-layout into WMMA B-fragment order (bf16) ----
// frag[w][kt][nt][lane][16 halves]; lanes 0-15: N=lane, K=kt*32+0..15
//                                   lanes16-31: N=lane-16, K=kt*32+16..31
__global__ void prep_wfrag(const float* __restrict__ Wq, const float* __restrict__ Wk,
                           const float* __restrict__ Wv, const float* __restrict__ Ws,
                           __bf16* __restrict__ frag) {
    int w = blockIdx.x, kt = blockIdx.y, nt = blockIdx.z;
    int lane = threadIdx.x;
    const float* W = (w == 0) ? Wq : (w == 1) ? Wk : (w == 2) ? Wv : Ws;
    int n  = nt * 16 + (lane & 15);
    int k0 = kt * 32 + ((lane < 16) ? 0 : 16);
    __bf16* dst = frag + ((((size_t)w * KT + kt) * NT + nt) * 32 + lane) * 16;
    #pragma unroll
    for (int j = 0; j < 16; ++j) {
        int k = k0 + j;
        float f = (k < D_DIM && n < H_DIM) ? W[k * H_DIM + n] : 0.0f;
        dst[j] = (__bf16)f;
    }
}

// ---- main: one workgroup per ego-graph ----
__global__ __launch_bounds__(128)
void graphenc_main(const int* __restrict__ nids, const int* __restrict__ adj,
                   const float* __restrict__ emb, const __bf16* __restrict__ frag,
                   const float* __restrict__ bq, const float* __restrict__ bk,
                   const float* __restrict__ bv, const float* __restrict__ bs,
                   float* __restrict__ out) {
    __shared__ struct __align__(16) {
        __bf16 x[64 * XS];        // gathered embeddings, bf16, zero-padded
        float  kbuf[64 * KVS];    // K projection (fp32)
        float  vbuf[64 * KVS];    // V projection (fp32)
        float  q0[112];           // q row 0 (bias included)
        float  sk0[112];          // skip row 0 (bias included)
        float  sc[64];            // masked scores
        float  wexp[64];          // softmax numerators
        int    nid[64];
    } S;

    const int tid  = threadIdx.x;
    const int lane = tid & 31;
    const int wave = tid >> 5;
    const int g    = blockIdx.x;

    // stage 0: neighbor ids
    if (tid < 64) S.nid[tid] = (tid < E_N) ? nids[(size_t)g * E_N + tid] : 0;
    __syncthreads();

    // stage 1: coalesced gather emb rows -> bf16 LDS, zero padding
    for (int row = wave; row < 64; row += 4) {
        const float* src = emb + (size_t)S.nid[row] * D_DIM;
        if (row + 4 < E_N)
            __builtin_prefetch(emb + (size_t)S.nid[row + 4] * D_DIM + lane, 0, 1);
        #pragma unroll
        for (int c = lane; c < XS; c += 32) {
            float f = (row < E_N && c < D_DIM) ? src[c] : 0.0f;
            S.x[row * XS + c] = (__bf16)f;
        }
    }
    __syncthreads();

    // stage 2: 70 WMMA tiles, round-robin across 4 waves (wave-uniform => EXEC all 1s)
    for (int t = wave; t < 70; t += 4) {
        int cls, mt, nt, w; const float* bias;
        if (t < 28)      { cls = 0; w = 1; bias = bk; mt = t / NT;        nt = t % NT; }
        else if (t < 56) { cls = 1; w = 2; bias = bv; mt = (t - 28) / NT; nt = (t - 28) % NT; }
        else if (t < 63) { cls = 2; w = 0; bias = bq; mt = 0;             nt = t - 56; }
        else             { cls = 3; w = 3; bias = bs; mt = 0;             nt = t - 63; }

        const int n    = nt * 16 + (lane & 15);
        const float bv0 = (n < H_DIM) ? bias[n] : 0.0f;
        v8f c;
        #pragma unroll
        for (int r = 0; r < 8; ++r) c[r] = bv0;   // bias folded into accumulator

        const int arow = (mt * 16 + (lane & 15)) * XS;
        #pragma unroll
        for (int kt = 0; kt < KT; ++kt) {
            const int kb = kt * 32 + ((lane < 16) ? 0 : 16);
            union { uint4 u[2]; v16bf v; } A, Bf;
            A.u[0] = *(const uint4*)&S.x[arow + kb];
            A.u[1] = *(const uint4*)&S.x[arow + kb + 8];
            const uint4* bp =
                (const uint4*)(frag + ((((size_t)w * KT + kt) * NT + nt) * 32 + lane) * 16);
            Bf.u[0] = bp[0];
            Bf.u[1] = bp[1];
            c = __builtin_amdgcn_wmma_f32_16x16x32_bf16(false, A.v, false, Bf.v,
                                                        (short)0, c, false, false);
        }

        if (cls <= 1) {                       // K / V: full 16x16 tile -> LDS fp32
            float* dst = (cls == 0) ? S.kbuf : S.vbuf;
            const int rbase = mt * 16 + ((lane < 16) ? 0 : 8);
            #pragma unroll
            for (int r = 0; r < 8; ++r) dst[(rbase + r) * KVS + n] = c[r];
        } else if (lane < 16) {               // Q / skip: only row M=0 (VGPR0, lanes 0-15)
            float* dst = (cls == 2) ? S.q0 : S.sk0;
            dst[n] = c[0];
        }
    }
    __syncthreads();

    // stage 3a: masked scores for target node 0:  mask[j] = adj[g, j, 0] != 0
    if (tid < E_N) {
        const int m = adj[((size_t)g * E_N + tid) * E_N];
        float s = 0.0f;
        for (int d = 0; d < D_DIM; ++d) s += S.q0[d] * S.kbuf[tid * KVS + d];
        S.sc[tid] = m ? s * 0.1f : -3.0e38f;  // 1/sqrt(H)=0.1
    }
    __syncthreads();

    // stage 3b: softmax numerators (self-loop guarantees a finite max)
    if (tid < E_N) {
        float mx = -3.0e38f;
        for (int j = 0; j < E_N; ++j) mx = fmaxf(mx, S.sc[j]);
        S.wexp[tid] = __expf(S.sc[tid] - mx);
    }
    __syncthreads();

    // stage 3c: aggregate + skip, write out[g, h]
    if (tid < H_DIM) {
        float den = 0.0f, acc = 0.0f;
        for (int j = 0; j < E_N; ++j) {
            const float wj = S.wexp[j];
            den += wj;
            acc += wj * S.vbuf[j * KVS + tid];
        }
        out[(size_t)g * H_DIM + tid] = acc / den + S.sk0[tid];
    }
}

extern "C" void kernel_launch(void* const* d_in, const int* in_sizes, int n_in,
                              void* d_out, int out_size, void* d_ws, size_t ws_size,
                              hipStream_t stream) {
    (void)in_sizes; (void)n_in; (void)out_size; (void)ws_size;
    const int*   nids = (const int*)d_in[0];
    const int*   adj  = (const int*)d_in[1];
    const float* emb  = (const float*)d_in[2];
    const float* Wq   = (const float*)d_in[3];
    const float* bq   = (const float*)d_in[4];
    const float* Wk   = (const float*)d_in[5];
    const float* bk   = (const float*)d_in[6];
    const float* Wv   = (const float*)d_in[7];
    const float* bv   = (const float*)d_in[8];
    const float* Ws   = (const float*)d_in[9];
    const float* bs   = (const float*)d_in[10];
    float*  out  = (float*)d_out;
    __bf16* frag = (__bf16*)d_ws;   // 4*4*7*32*16 bf16 = 114688 bytes

    prep_wfrag<<<dim3(4, KT, NT), 32, 0, stream>>>(Wq, Wk, Wv, Ws, frag);
    graphenc_main<<<GCOUNT, 128, 0, stream>>>(nids, adj, emb, frag,
                                              bq, bk, bv, bs, out);
}
